// MultiHeadAttention_35605278883906
// MI455X (gfx1250) — matrix operand order repaired and checked by reading, not compiled, not run on hardware
//
#include <hip/hip_runtime.h>

// MI455X (gfx1250) multi-head attention forward.
// f16 WMMA everywhere (v_wmma_f32_16x16x32_f16), f32 accumulate.
// Flash-attention inner loop: scores never hit HBM (saves 2x1.6GB traffic).
// All f16 operands live in the 192MB L2; GEMMs stream fragments directly.

typedef _Float16 f16;
typedef __attribute__((ext_vector_type(16))) _Float16 v16h;
typedef __attribute__((ext_vector_type(8)))  _Float16 v8h;
typedef __attribute__((ext_vector_type(8)))  float    v8f;

namespace {
constexpr int kD  = 768;
constexpr int kH  = 12;
constexpr int kDk = 64;
constexpr int kB  = 2;
constexpr int kS  = 4096;
constexpr int kM  = kB * kS;  // 8192 tokens
}

__device__ __forceinline__ v8f wmma_f16f32(v16h a, v16h b, v8f c) {
  // (neg_a, A, neg_b, B, c_mod, C, reuse_a, reuse_b)
  return __builtin_amdgcn_wmma_f32_16x16x32_f16(false, a, false, b, (short)0, c,
                                                false, false);
}

// A-fragment (16x32, M x K), source row-major with leading dim ld (f16 elems).
// Lane L: M = L&15, half = L>>4; a[i] = A[M][half*16+i], a[8+i] = A[M][half*16+8+i].
__device__ __forceinline__ v16h load_A16(const f16* __restrict__ base, int ld,
                                         int m0, int k0, int lane) {
  const int M = lane & 15, h = lane >> 4;
  const f16* p = base + (size_t)(m0 + M) * ld + k0 + h * 16;
  v8h lo = *(const v8h*)(p);
  v8h hi = *(const v8h*)(p + 8);
  v16h a;
#pragma unroll
  for (int i = 0; i < 8; ++i) { a[i] = lo[i]; a[i + 8] = hi[i]; }
  return a;
}

// B-fragment (32x16, K x N) where B[k][n] = SRC[n][k] (SRC row-major, ld elems).
// Lane L: N = L&15, half = L>>4; b[j] = SRC[n0+N][k0 + half*16 + j]  (contiguous!)
__device__ __forceinline__ v16h load_Bt16(const f16* __restrict__ base, int ld,
                                          int n0, int k0, int lane) {
  const int N = lane & 15, h = lane >> 4;
  const f16* p = base + (size_t)(n0 + N) * ld + k0 + h * 16;
  v8h lo = *(const v8h*)(p);
  v8h hi = *(const v8h*)(p + 8);
  v16h b;
#pragma unroll
  for (int i = 0; i < 8; ++i) { b[i] = lo[i]; b[i + 8] = hi[i]; }
  return b;
}

__global__ void mha_cvt_f16(const float* __restrict__ in, f16* __restrict__ out,
                            int n) {
  int i = blockIdx.x * blockDim.x + threadIdx.x;
  if (i < n) out[i] = (f16)in[i];
}

// C[m][n] = sum_k X[m][k] * W[n][k] + bias[n]   (x @ W.T + b), M=8192 N=K=768.
// Block: 256 thr = 8 waves (2x4); wave tile 32x64 (2x4 WMMA accs); block 64x256.
// 6 fragment loads feed 8 WMMAs per k-step (0.75 frags/WMMA from L2).
// mode 0: Q -> [B,H,S,64] f16, scaled by 1/sqrt(dk)
// mode 1: K -> [B,H,S,64] f16
// mode 2: V -> [B,H,64,S] f16 (transposed for PV B-fragments)
// mode 3: O -> [M,768] f32 (final output)
__global__ void mha_gemm_wmma(const f16* __restrict__ X, const f16* __restrict__ W,
                              const float* __restrict__ bias, void* __restrict__ outp,
                              int mode, float scale) {
  const int wid = threadIdx.x >> 5;
  const int lane = threadIdx.x & 31;
  const int wy = wid >> 2, wx = wid & 3;
  const int m0 = blockIdx.y * 64 + wy * 32;
  const int n0 = blockIdx.x * 256 + wx * 64;

  v8f acc[2][4] = {};
  for (int k0 = 0; k0 < kD; k0 += 32) {
    v16h a0 = load_A16(X, kD, m0, k0, lane);
    v16h a1 = load_A16(X, kD, m0 + 16, k0, lane);
#pragma unroll
    for (int j = 0; j < 4; ++j) {
      v16h bj = load_Bt16(W, kD, n0 + j * 16, k0, lane);
      acc[0][j] = wmma_f16f32(a0, bj, acc[0][j]);
      acc[1][j] = wmma_f16f32(a1, bj, acc[1][j]);
    }
  }

  const int half = lane >> 4;
  const int N = lane & 15;
  const int rowbase = half * 8;
#pragma unroll
  for (int i = 0; i < 2; ++i) {
#pragma unroll
    for (int j = 0; j < 4; ++j) {
#pragma unroll
      for (int r = 0; r < 8; ++r) {
        const int m = m0 + i * 16 + rowbase + r;
        const int n = n0 + j * 16 + N;
        float val = acc[i][j][r] + bias[n];
        if (mode == 3) {
          ((float*)outp)[(size_t)m * kD + n] = val;
        } else {
          const int bi = m >> 12;         // m / S
          const int s  = m & (kS - 1);
          const int h  = n >> 6;          // n / dk
          const int d  = n & (kDk - 1);
          f16* o16 = (f16*)outp;
          if (mode == 2) {                // V transposed: [B,H,dk,S]
            o16[((size_t)(bi * kH + h) * kDk + d) * kS + s] = (f16)val;
          } else {                        // Q/K: [B,H,S,dk]
            o16[((size_t)(bi * kH + h) * kS + s) * kDk + d] = (f16)(val * scale);
          }
        }
      }
    }
  }
}

// One wave per (b, h, 16-row q-tile); 8 waves/block. Online-softmax flash attn.
// 64-key chunks: 8 WMMAs (Q K^T) + one softmax update + LDS C->A relayout +
// 8 WMMAs (P V). Shuffle/rescale bookkeeping amortized over 64 keys.
__global__ void mha_flash_wmma(const f16* __restrict__ Q, const f16* __restrict__ K,
                               const f16* __restrict__ Vt, f16* __restrict__ Aout) {
  __shared__ f16 plds[8][16 * 64];  // 2KB P-tile per wave
  const int wid = threadIdx.x >> 5;
  const int lane = threadIdx.x & 31;
  const int w = blockIdx.x * 8 + wid;
  const int qt = w & 255;           // S/16 = 256 q tiles
  const int rest = w >> 8;
  const int h = rest % kH;
  const int b = rest / kH;
  const int q0 = qt * 16;

  const f16* Qbh = Q + (size_t)(b * kH + h) * kS * kDk;
  const f16* Kbh = K + (size_t)(b * kH + h) * kS * kDk;
  const f16* Vbh = Vt + (size_t)(b * kH + h) * kDk * kS;

  // Q tile (16 x 64), already pre-scaled by 1/sqrt(dk): two A-fragments, resident.
  const v16h qa0 = load_A16(Qbh, kDk, q0, 0, lane);
  const v16h qa1 = load_A16(Qbh, kDk, q0, 32, lane);

  const int half = lane >> 4;
  const int N = lane & 15;
  const int rowbase = half * 8;   // C/D layout: this lane holds rows rowbase..rowbase+7

  float m_r[8], l_r[8];
#pragma unroll
  for (int r = 0; r < 8; ++r) { m_r[r] = -1e30f; l_r[r] = 0.f; }
  v8f o[4] = {};

  for (int kc = 0; kc < kS / 64; ++kc) {
    const int key0 = kc * 64;
    // scores tile S(16x64) = Q(16x64) @ K^T(64x64): four 16x16 accumulators
    v8f sc[4] = {};
#pragma unroll
    for (int t = 0; t < 4; ++t) {
      v16h bk0 = load_Bt16(Kbh, kDk, key0 + t * 16, 0, lane);
      v16h bk1 = load_Bt16(Kbh, kDk, key0 + t * 16, 32, lane);
      sc[t] = wmma_f16f32(qa0, bk0, sc[t]);
      sc[t] = wmma_f16f32(qa1, bk1, sc[t]);
    }

    // Online softmax. Rows live across a 16-lane half-wave -> xor masks 1..8.
    float alpha[8], p[4][8];
#pragma unroll
    for (int r = 0; r < 8; ++r) {
      float vmx = fmaxf(fmaxf(sc[0][r], sc[1][r]), fmaxf(sc[2][r], sc[3][r]));
      vmx = fmaxf(vmx, __shfl_xor(vmx, 1, 32));
      vmx = fmaxf(vmx, __shfl_xor(vmx, 2, 32));
      vmx = fmaxf(vmx, __shfl_xor(vmx, 4, 32));
      vmx = fmaxf(vmx, __shfl_xor(vmx, 8, 32));
      const float mn = fmaxf(m_r[r], vmx);
      alpha[r] = __expf(m_r[r] - mn);
      m_r[r] = mn;
      float ps = 0.f;
#pragma unroll
      for (int t = 0; t < 4; ++t) {
        p[t][r] = __expf(sc[t][r] - mn);
        ps += p[t][r];
      }
      ps += __shfl_xor(ps, 1, 32);
      ps += __shfl_xor(ps, 2, 32);
      ps += __shfl_xor(ps, 4, 32);
      ps += __shfl_xor(ps, 8, 32);
      l_r[r] = l_r[r] * alpha[r] + ps;
    }
#pragma unroll
    for (int t = 0; t < 4; ++t)
#pragma unroll
      for (int r = 0; r < 8; ++r) o[t][r] *= alpha[r];

    // Relayout P (C-layout -> A-fragments) through wave-private LDS.
    f16* pl = &plds[wid][0];
    asm volatile("" ::: "memory");
#pragma unroll
    for (int t = 0; t < 4; ++t)
#pragma unroll
      for (int r = 0; r < 8; ++r)
        pl[(rowbase + r) * 64 + t * 16 + N] = (f16)p[t][r];
    asm volatile("s_wait_dscnt 0x0" ::: "memory");
    const v16h pa0 = load_A16(pl, 64, 0, 0, lane);
    const v16h pa1 = load_A16(pl, 64, 0, 32, lane);
    asm volatile("s_wait_dscnt 0x0" ::: "memory");

    // O(16x64) += P(16x64) @ V(64x64); Vt is dk-major so B-frags are contiguous.
#pragma unroll
    for (int t = 0; t < 4; ++t) {
      v16h bv0 = load_Bt16(Vbh, kS, t * 16, key0, lane);
      v16h bv1 = load_Bt16(Vbh, kS, t * 16, key0 + 32, lane);
      o[t] = wmma_f16f32(pa0, bv0, o[t]);
      o[t] = wmma_f16f32(pa1, bv1, o[t]);
    }
  }

  // Normalize and write [B,S,H*dk] f16 for the output projection.
#pragma unroll
  for (int t = 0; t < 4; ++t) {
#pragma unroll
    for (int r = 0; r < 8; ++r) {
      const float val = o[t][r] / l_r[r];
      const int tok = q0 + rowbase + r;
      const int col = h * kDk + t * 16 + N;
      Aout[(size_t)(b * kS + tok) * kD + col] = (f16)val;
    }
  }
}

extern "C" void kernel_launch(void* const* d_in, const int* in_sizes, int n_in,
                              void* d_out, int out_size, void* d_ws, size_t ws_size,
                              hipStream_t stream) {
  const float* q  = (const float*)d_in[0];
  const float* k  = (const float*)d_in[1];
  const float* v  = (const float*)d_in[2];
  const float* Wq = (const float*)d_in[3];
  const float* bq = (const float*)d_in[4];
  const float* Wk = (const float*)d_in[5];
  const float* bk = (const float*)d_in[6];
  const float* Wv = (const float*)d_in[7];
  const float* bv = (const float*)d_in[8];
  const float* Wo = (const float*)d_in[9];
  const float* bo = (const float*)d_in[10];

  char* ws = (char*)d_ws;
  size_t off = 0;
  auto carve = [&](size_t elems) -> f16* {
    f16* p = (f16*)(ws + off);
    off = (off + elems * sizeof(f16) + 255) & ~(size_t)255;
    return p;
  };
  const int nAct = kM * kD;        // 6291456
  const int nW   = kD * kD;        // 589824
  f16* Xq16 = carve(nAct);
  f16* Xk16 = carve(nAct);
  f16* Xv16 = carve(nAct);
  f16* Wq16 = carve(nW);
  f16* Wk16 = carve(nW);
  f16* Wv16 = carve(nW);
  f16* Wo16 = carve(nW);
  f16* Q16  = carve((size_t)kB * kH * kS * kDk);
  f16* K16  = carve((size_t)kB * kH * kS * kDk);
  f16* Vt16 = carve((size_t)kB * kH * kDk * kS);
  f16* A16  = Xq16;  // alias: Xq16 is dead after the Q projection

  // f32 -> f16 converts
  mha_cvt_f16<<<(nAct + 255) / 256, 256, 0, stream>>>(q, Xq16, nAct);
  mha_cvt_f16<<<(nAct + 255) / 256, 256, 0, stream>>>(k, Xk16, nAct);
  mha_cvt_f16<<<(nAct + 255) / 256, 256, 0, stream>>>(v, Xv16, nAct);
  mha_cvt_f16<<<(nW + 255) / 256, 256, 0, stream>>>(Wq, Wq16, nW);
  mha_cvt_f16<<<(nW + 255) / 256, 256, 0, stream>>>(Wk, Wk16, nW);
  mha_cvt_f16<<<(nW + 255) / 256, 256, 0, stream>>>(Wv, Wv16, nW);
  mha_cvt_f16<<<(nW + 255) / 256, 256, 0, stream>>>(Wo, Wo16, nW);

  // Projections (block tile 64x256, grid 3x128)
  dim3 ggrid(kD / 256, kM / 64);
  dim3 gblk(256);
  mha_gemm_wmma<<<ggrid, gblk, 0, stream>>>(Xq16, Wq16, bq, (void*)Q16, 0, 0.125f);
  mha_gemm_wmma<<<ggrid, gblk, 0, stream>>>(Xk16, Wk16, bk, (void*)K16, 1, 1.0f);
  mha_gemm_wmma<<<ggrid, gblk, 0, stream>>>(Xv16, Wv16, bv, (void*)Vt16, 2, 1.0f);

  // Flash attention: 2*12*256 q-tiles = 6144 waves = 768 blocks of 8 waves
  mha_flash_wmma<<<dim3((kB * kH * (kS / 16)) / 8), dim3(256), 0, stream>>>(
      Q16, K16, Vt16, A16);

  // Output projection -> fp32 d_out
  mha_gemm_wmma<<<ggrid, gblk, 0, stream>>>(A16, Wo16, bo, d_out, 3, 1.0f);
}
